// CausalSelfAttention_75728863363576
// MI455X (gfx1250) — matrix operand order repaired and checked by reading, not compiled, not run on hardware
//
#include <hip/hip_runtime.h>
#include <hip/hip_bf16.h>

// ---------------------------------------------------------------------------
// CDNA5 (gfx1250, wave32) fused GQA sliding-window attention
//   stage 0: x -> bf16, weights -> bf16 repacked into WMMA B-operand tiles
//   stage 1: q = x@Wq^T, kv = x@Wkv^T   (v_wmma_f32_16x16x32_bf16, f32 out)
//   stage 2: QK L2-norm + RoPE -> bf16 head-major; V -> bf16 transposed
//   stage 3: flash attention per (head, 16-query tile), window = 512
//   stage 4: out = y@Wproj^T (WMMA, f32 out to d_out)
// ---------------------------------------------------------------------------

typedef __attribute__((ext_vector_type(16))) __bf16 bf16x16;
typedef __attribute__((ext_vector_type(8)))  __bf16 bf16x8;
typedef __attribute__((ext_vector_type(8)))  float  f32x8;

#define T_SEQ  2048
#define C_DIM  1024
#define NH     16
#define NKV    4
#define HD     64
#define WIN    512
#define KV_OUT 512

static __device__ __forceinline__ bf16x16 ld_pair(const __bf16* p0, const __bf16* p1) {
  union { bf16x16 v; bf16x8 h[2]; } u;
  u.h[0] = *(const bf16x8*)p0;
  u.h[1] = *(const bf16x8*)p1;
  return u.v;
}

static __device__ __forceinline__ f32x8 wmma_bf16(bf16x16 a, bf16x16 b, f32x8 c) {
  // D = A x B + C : v_wmma_f32_16x16x32_bf16
  return __builtin_amdgcn_wmma_f32_16x16x32_bf16(false, a, false, b, (short)0, c,
                                                 false, false);
}

// ---------------- stage 0: conversions / weight repack ----------------------

__global__ void cvt_f32_bf16(const float* __restrict__ src, __bf16* __restrict__ dst, int n) {
  int i = blockIdx.x * blockDim.x + threadIdx.x;
  if (i < n) dst[i] = (__bf16)src[i];
}

// Repack W[N][K] (f32 row-major) into per-(16x32)-tile lane-major bf16 blocks:
// block b = ntile*(K/32)+ktile, lane L holds B-operand col n=L%16,
// K rows (L/16)*8+e (e<8), 16+(L/16)*8+e-8 (e>=8) -> 32 contiguous bytes per lane.
__global__ void repack_w(const float* __restrict__ W, __bf16* __restrict__ Wr, int N, int K) {
  int idx = blockIdx.x * blockDim.x + threadIdx.x;
  if (idx >= N * K) return;
  int e  = idx & 15;
  int L  = (idx >> 4) & 31;
  int b  = idx >> 9;
  int kb = K >> 5;
  int kt = b % kb;
  int nt = b / kb;
  int n  = nt * 16 + (L & 15);
  int k  = kt * 32 + (e < 8 ? (L >> 4) * 8 + e : 16 + (L >> 4) * 8 + (e - 8));
  Wr[idx] = (__bf16)W[(size_t)n * K + k];
}

// ---------------- stage 1/4: WMMA GEMM  C[M][N] = A[M][K] * Wr^T ------------
// One wave computes a 32(M) x 64(N) strip: 8 accumulators, A reused 4x,
// B reused 2x. Next k-step operands are loaded before the current step's
// WMMA chain so VMEM overlaps the matrix pipe; last k-step is peeled so the
// inner loop is a single fallthrough (loads -> 8x WMMA). global_prefetch
// warms GL2 ~4 k-steps ahead on the streaming operands.

__global__ void __launch_bounds__(128) gemm_bf16(const __bf16* __restrict__ A,
                                                 const __bf16* __restrict__ Br,
                                                 float* __restrict__ Cout,
                                                 int M, int N, int K) {
  int wid  = (blockIdx.x * blockDim.x + threadIdx.x) >> 5;
  int lane = threadIdx.x & 31;
  int nb64 = N >> 6;
  int mt   = wid / nb64;
  int nb   = wid % nb64;
  if (mt >= (M >> 5)) return;
  int hlf = lane >> 4, l16 = lane & 15;
  int kblocks = K >> 5;

  const __bf16* arow0 = A + (size_t)(mt * 32 + l16) * K + hlf * 8;
  const __bf16* arow1 = arow0 + (size_t)16 * K;
  const __bf16* bp0 = Br + (size_t)(nb * 4 + 0) * kblocks * 512 + lane * 16;
  const __bf16* bp1 = Br + (size_t)(nb * 4 + 1) * kblocks * 512 + lane * 16;
  const __bf16* bp2 = Br + (size_t)(nb * 4 + 2) * kblocks * 512 + lane * 16;
  const __bf16* bp3 = Br + (size_t)(nb * 4 + 3) * kblocks * 512 + lane * 16;

  f32x8 acc00 = {}, acc01 = {}, acc02 = {}, acc03 = {};
  f32x8 acc10 = {}, acc11 = {}, acc12 = {}, acc13 = {};

  // prologue: operands for k-step 0
  bf16x16 a0 = ld_pair(arow0, arow0 + 16);
  bf16x16 a1 = ld_pair(arow1, arow1 + 16);
  bf16x16 b0 = ld_pair(bp0, bp0 + 8);
  bf16x16 b1 = ld_pair(bp1, bp1 + 8);
  bf16x16 b2 = ld_pair(bp2, bp2 + 8);
  bf16x16 b3 = ld_pair(bp3, bp3 + 8);

  for (int kt = 0; kt + 1 < kblocks; ++kt) {
    int k0 = (kt + 1) * 32;
    int bo = (kt + 1) * 512;
    // prefetch ~4 k-steps ahead (sequential streams; harmless past the end)
    __builtin_prefetch(arow0 + k0 + 96, 0, 1);
    __builtin_prefetch(bp0 + bo + 1536, 0, 1);
    __builtin_prefetch(bp2 + bo + 1536, 0, 1);
    // next k-step operands (issue before the WMMA chain)
    bf16x16 na0 = ld_pair(arow0 + k0, arow0 + k0 + 16);
    bf16x16 na1 = ld_pair(arow1 + k0, arow1 + k0 + 16);
    bf16x16 nb0 = ld_pair(bp0 + bo, bp0 + bo + 8);
    bf16x16 nb1 = ld_pair(bp1 + bo, bp1 + bo + 8);
    bf16x16 nb2 = ld_pair(bp2 + bo, bp2 + bo + 8);
    bf16x16 nb3 = ld_pair(bp3 + bo, bp3 + bo + 8);

    acc00 = wmma_bf16(a0, b0, acc00);
    acc01 = wmma_bf16(a0, b1, acc01);
    acc02 = wmma_bf16(a0, b2, acc02);
    acc03 = wmma_bf16(a0, b3, acc03);
    acc10 = wmma_bf16(a1, b0, acc10);
    acc11 = wmma_bf16(a1, b1, acc11);
    acc12 = wmma_bf16(a1, b2, acc12);
    acc13 = wmma_bf16(a1, b3, acc13);

    a0 = na0; a1 = na1;
    b0 = nb0; b1 = nb1; b2 = nb2; b3 = nb3;
  }
  // peeled final k-step
  acc00 = wmma_bf16(a0, b0, acc00);
  acc01 = wmma_bf16(a0, b1, acc01);
  acc02 = wmma_bf16(a0, b2, acc02);
  acc03 = wmma_bf16(a0, b3, acc03);
  acc10 = wmma_bf16(a1, b0, acc10);
  acc11 = wmma_bf16(a1, b1, acc11);
  acc12 = wmma_bf16(a1, b2, acc12);
  acc13 = wmma_bf16(a1, b3, acc13);

  int ncol0 = nb * 64 + l16;
  int rowb0 = mt * 32 + hlf * 8;
  int rowb1 = rowb0 + 16;
#pragma unroll
  for (int r = 0; r < 8; ++r) {
    float* c0 = Cout + (size_t)(rowb0 + r) * N + ncol0;
    c0[0]  = acc00[r];
    c0[16] = acc01[r];
    c0[32] = acc02[r];
    c0[48] = acc03[r];
    float* c1 = Cout + (size_t)(rowb1 + r) * N + ncol0;
    c1[0]  = acc10[r];
    c1[16] = acc11[r];
    c1[32] = acc12[r];
    c1[48] = acc13[r];
  }
}

// ---------------- stage 2: QK L2-norm + RoPE -> bf16 head-major -------------
// One wave per (head, t): 64 elements, 2 per lane (i=lane, i+32).

__global__ void __launch_bounds__(128) qk_norm_rope(const float* __restrict__ src, int stride,
                                                    int nh, __bf16* __restrict__ dst) {
  int w    = (blockIdx.x * blockDim.x + threadIdx.x) >> 5;
  int lane = threadIdx.x & 31;
  int t = w % T_SEQ;
  int h = w / T_SEQ;
  if (h >= nh) return;
  const float* p = src + (size_t)t * stride + h * HD;
  float v0 = p[lane], v1 = p[lane + 32];
  float ss = v0 * v0 + v1 * v1;
#pragma unroll
  for (int m = 16; m >= 1; m >>= 1) ss += __shfl_xor(ss, m, 32);
  float inv = 1.0f / (sqrtf(ss) + 1e-6f);
  v0 *= inv; v1 *= inv;
  // theta_j = 10000^(-j/32), f = t * theta_lane
  float f = (float)t * __expf(-(float)lane * 0.28782313662425574f);
  float s, c;
  __sincosf(f, &s, &c);
  float o0 = v0 * c - v1 * s;   // rotate_half: [-t2, t1]
  float o1 = v1 * c + v0 * s;
  __bf16* q = dst + ((size_t)h * T_SEQ + t) * HD;
  q[lane]      = (__bf16)o0;
  q[lane + 32] = (__bf16)o1;
}

// V: kvf[t][256 + kvh*64 + i] -> vtb[kvh][i][t] (bf16, transposed for B-operand)
__global__ void v_transpose(const float* __restrict__ kvf, __bf16* __restrict__ vtb) {
  int idx = blockIdx.x * blockDim.x + threadIdx.x;
  if (idx >= NKV * T_SEQ * HD) return;
  int i   = idx & 63;
  int t   = (idx >> 6) & (T_SEQ - 1);
  int kvh = idx >> 17;
  vtb[((size_t)kvh * HD + i) * T_SEQ + t] =
      (__bf16)kvf[(size_t)t * KV_OUT + 256 + kvh * HD + i];
}

// ---------------- stage 3: flash attention (window 512) ---------------------
// One wave per (head h, 16-query tile). Key tiles processed in pairs (K=32
// for the P*V WMMA); P converted C-layout -> A-layout via an LDS round trip.

__global__ void __launch_bounds__(128) attn_kernel(const __bf16* __restrict__ qb,
                                                   const __bf16* __restrict__ kmat,
                                                   const __bf16* __restrict__ vtb,
                                                   __bf16* __restrict__ yb) {
  __shared__ __bf16 pstage[4][16 * 32];
  int wid  = threadIdx.x >> 5;
  int lane = threadIdx.x & 31;
  int gw   = blockIdx.x * 4 + wid;
  int qt = gw % (T_SEQ / 16);
  int h  = gw / (T_SEQ / 16);
  if (h >= NH) return;
  int hlf = lane >> 4, l16 = lane & 15;
  int q0  = qt * 16;
  int kv  = h >> 2;                       // n_rep = 4 (repeat_interleave)

  const __bf16* qrow = qb + ((size_t)h * T_SEQ + q0 + l16) * HD + hlf * 16;
  bf16x16 qa0 = ld_pair(qrow,      qrow + 8);
  bf16x16 qa1 = ld_pair(qrow + 32, qrow + 40);

  f32x8 acc0 = {}, acc1 = {}, acc2 = {}, acc3 = {};
  float m_i[8], l_i[8];
  const float NEG = -__builtin_inff();
#pragma unroll
  for (int r = 0; r < 8; ++r) { m_i[r] = NEG; l_i[r] = 0.f; }

  int kt_lo = (q0 - WIN + 1) > 0 ? (q0 - WIN + 1) >> 4 : 0;
  int ktE   = kt_lo & ~1;                 // even start; extra cols get masked
  int kt_hi = qt;
  __bf16* pl = &pstage[wid][0];
  const float scale = 0.125f;             // 1/sqrt(64)

  for (int kt2 = ktE; kt2 <= kt_hi; kt2 += 2) {
    int kt1 = kt2 + 1;                    // always <= 127 (in-bounds loads)
    // --- S = Q K^T for the two 16-key tiles ---
    f32x8 s0 = {}, s1 = {};
    {
      const __bf16* kr = kmat + ((size_t)kv * T_SEQ + kt2 * 16 + l16) * HD + hlf * 16;
      s0 = wmma_bf16(qa0, ld_pair(kr, kr + 8), s0);
      s0 = wmma_bf16(qa1, ld_pair(kr + 32, kr + 40), s0);
    }
    {
      const __bf16* kr = kmat + ((size_t)kv * T_SEQ + kt1 * 16 + l16) * HD + hlf * 16;
      s1 = wmma_bf16(qa0, ld_pair(kr, kr + 8), s1);
      s1 = wmma_bf16(qa1, ld_pair(kr + 32, kr + 40), s1);
    }
    int col0 = kt2 * 16 + l16;
    int col1 = kt1 * 16 + l16;
    bool t1v = kt1 <= kt_hi;

    // --- online softmax + stage P into LDS (C-layout -> A-layout) ---
#pragma unroll
    for (int r = 0; r < 8; ++r) {
      int row = q0 + r + 8 * hlf;
      float a = s0[r] * scale, b = s1[r] * scale;
      bool va = (col0 <= row) && (col0 + WIN > row);
      bool vb = t1v && (col1 <= row) && (col1 + WIN > row);
      a = va ? a : NEG;
      b = vb ? b : NEG;
      float tm = fmaxf(a, b);
#pragma unroll
      for (int m = 8; m >= 1; m >>= 1) tm = fmaxf(tm, __shfl_xor(tm, m, 32));
      float mnew  = fmaxf(m_i[r], tm);
      float alpha = (m_i[r] == mnew) ? 1.0f : __expf(m_i[r] - mnew);
      float pa = va ? __expf(a - mnew) : 0.0f;
      float pb = vb ? __expf(b - mnew) : 0.0f;
      float rs = pa + pb;
#pragma unroll
      for (int m = 8; m >= 1; m >>= 1) rs += __shfl_xor(rs, m, 32);
      l_i[r] = l_i[r] * alpha + rs;
      m_i[r] = mnew;
      acc0[r] *= alpha; acc1[r] *= alpha; acc2[r] *= alpha; acc3[r] *= alpha;
      pl[(r + 8 * hlf) * 32 + l16]      = (__bf16)pa;
      pl[(r + 8 * hlf) * 32 + 16 + l16] = (__bf16)pb;
    }

    // --- reload P as A-operand, accumulate P*V (K = 32 keys) ---
    const __bf16* prow = pl + l16 * 32 + hlf * 16;
    bf16x16 pa16 = ld_pair(prow, prow + 8);
    {
      const __bf16* vc = vtb + ((size_t)kv * HD + 0 * 16 + l16) * T_SEQ + kt2 * 16 + hlf * 16;
      acc0 = wmma_bf16(pa16, ld_pair(vc, vc + 8), acc0);
    }
    {
      const __bf16* vc = vtb + ((size_t)kv * HD + 1 * 16 + l16) * T_SEQ + kt2 * 16 + hlf * 16;
      acc1 = wmma_bf16(pa16, ld_pair(vc, vc + 8), acc1);
    }
    {
      const __bf16* vc = vtb + ((size_t)kv * HD + 2 * 16 + l16) * T_SEQ + kt2 * 16 + hlf * 16;
      acc2 = wmma_bf16(pa16, ld_pair(vc, vc + 8), acc2);
    }
    {
      const __bf16* vc = vtb + ((size_t)kv * HD + 3 * 16 + l16) * T_SEQ + kt2 * 16 + hlf * 16;
      acc3 = wmma_bf16(pa16, ld_pair(vc, vc + 8), acc3);
    }
  }

  // --- epilogue: divide by row sums, store bf16 row-major [T][C] ---
#pragma unroll
  for (int r = 0; r < 8; ++r) {
    float invl = 1.0f / l_i[r];
    int row = q0 + r + 8 * hlf;
    __bf16* orow = yb + (size_t)row * C_DIM + h * HD + l16;
    orow[0]  = (__bf16)(acc0[r] * invl);
    orow[16] = (__bf16)(acc1[r] * invl);
    orow[32] = (__bf16)(acc2[r] * invl);
    orow[48] = (__bf16)(acc3[r] * invl);
  }
}

// ---------------------------------------------------------------------------

extern "C" void kernel_launch(void* const* d_in, const int* in_sizes, int n_in,
                              void* d_out, int out_size, void* d_ws, size_t ws_size,
                              hipStream_t stream) {
  (void)in_sizes; (void)n_in; (void)out_size; (void)ws_size;
  const float* x   = (const float*)d_in[0];
  const float* Wq  = (const float*)d_in[1];
  const float* Wkv = (const float*)d_in[2];
  const float* Wp  = (const float*)d_in[3];

  char* ws = (char*)d_ws;
  size_t off = 0;
  auto alloc = [&](size_t bytes) -> void* {
    void* p = ws + off;
    off += (bytes + 255) & ~(size_t)255;
    return p;
  };

  __bf16* xb    = (__bf16*)alloc((size_t)T_SEQ * C_DIM * 2);
  __bf16* Wq_r  = (__bf16*)alloc((size_t)C_DIM * C_DIM * 2);
  __bf16* Wkv_r = (__bf16*)alloc((size_t)KV_OUT * C_DIM * 2);
  __bf16* Wp_r  = (__bf16*)alloc((size_t)C_DIM * C_DIM * 2);
  float*  qf    = (float*) alloc((size_t)T_SEQ * C_DIM * 4);
  float*  kvf   = (float*) alloc((size_t)T_SEQ * KV_OUT * 4);
  __bf16* qb    = (__bf16*)alloc((size_t)NH  * T_SEQ * HD * 2);
  __bf16* kbuf  = (__bf16*)alloc((size_t)NKV * T_SEQ * HD * 2);
  __bf16* vtb   = (__bf16*)alloc((size_t)NKV * HD * T_SEQ * 2);
  __bf16* yb    = (__bf16*)alloc((size_t)T_SEQ * C_DIM * 2);

  // stage 0
  cvt_f32_bf16<<<(T_SEQ * C_DIM) / 256, 256, 0, stream>>>(x, xb, T_SEQ * C_DIM);
  repack_w<<<(C_DIM * C_DIM) / 256, 256, 0, stream>>>(Wq, Wq_r, C_DIM, C_DIM);
  repack_w<<<(KV_OUT * C_DIM) / 256, 256, 0, stream>>>(Wkv, Wkv_r, KV_OUT, C_DIM);
  repack_w<<<(C_DIM * C_DIM) / 256, 256, 0, stream>>>(Wp, Wp_r, C_DIM, C_DIM);

  // stage 1: projections (waves = (M/32)*(N/64), 4 waves/block)
  gemm_bf16<<<(T_SEQ / 32) * (C_DIM / 64) / 4, 128, 0, stream>>>(xb, Wq_r, qf,
                                                                 T_SEQ, C_DIM, C_DIM);
  gemm_bf16<<<(T_SEQ / 32) * (KV_OUT / 64) / 4, 128, 0, stream>>>(xb, Wkv_r, kvf,
                                                                  T_SEQ, KV_OUT, C_DIM);

  // stage 2: norm + rope (q from qf; k = first 256 cols of kvf), v transpose
  qk_norm_rope<<<(NH * T_SEQ) / 4, 128, 0, stream>>>(qf, C_DIM, NH, qb);
  qk_norm_rope<<<(NKV * T_SEQ) / 4, 128, 0, stream>>>(kvf, KV_OUT, NKV, kbuf);
  v_transpose<<<(NKV * T_SEQ * HD) / 256, 256, 0, stream>>>(kvf, vtb);

  // stage 3: attention (one wave per (head, q-tile))
  attn_kernel<<<(NH * (T_SEQ / 16)) / 4, 128, 0, stream>>>(qb, kbuf, vtb, yb);

  // stage 4: output projection -> d_out (f32)
  gemm_bf16<<<(T_SEQ / 32) * (C_DIM / 64) / 4, 128, 0, stream>>>(yb, Wp_r, (float*)d_out,
                                                                 T_SEQ, C_DIM, C_DIM);
}
